// MambaBlock_17849884082758
// MI455X (gfx1250) — matrix operand order repaired and checked by reading, not compiled, not run on hardware
//
#include <hip/hip_runtime.h>
#include <hip/hip_bf16.h>
#include <stdint.h>

// ---------------------------------------------------------------------------
// Mamba (bimamba v2) block for MI455X / gfx1250.
// Heavy GEMMs: v_wmma_f32_16x16x32_bf16 with double-buffered LDS tiles fed by
// GLOBAL_LOAD_ASYNC_TO_LDS_B128 (ASYNCcnt) when available. Selective scan is
// f32 VALU with register-resident state.
// ---------------------------------------------------------------------------

#define D_MODEL   768
#define D_STATE   16
#define D_CONV    4
#define D_INNER   1536
#define DT_RANK   48
#define DT_PAD    64
#define BATCH     4
#define SEQ       1024
#define ROWS      4096      // BATCH * SEQ
#define XDBL_C    80        // DT_RANK + 2*D_STATE

typedef __attribute__((ext_vector_type(16))) __bf16 v16bf;
typedef __attribute__((ext_vector_type(8)))  float  v8f;
typedef __attribute__((__vector_size__(4 * sizeof(int)))) int v4i_t;

#if defined(__has_builtin)
#if __has_builtin(__builtin_amdgcn_global_load_async_to_lds_b128) && \
    __has_builtin(__builtin_amdgcn_s_wait_asynccnt)
#define USE_ASYNC_LDS 1
#endif
#endif

static __device__ __forceinline__ unsigned short f2bf(float f) {
  unsigned int u = __float_as_uint(f);
  if ((u & 0x7fffffffu) > 0x7f800000u)             // NaN -> quiet
    return (unsigned short)((u >> 16) | 0x40u);
  unsigned int r = u + 0x7fffu + ((u >> 16) & 1u); // round-to-nearest-even
  return (unsigned short)(r >> 16);
}
static __device__ __forceinline__ float bf2f(unsigned short s) {
  return __uint_as_float(((unsigned int)s) << 16);
}
static __device__ __forceinline__ float silu_f(float x) {
  return x / (1.0f + __expf(-x));
}
static __device__ __forceinline__ v8f v8f_zero() {
  v8f z;
#pragma unroll
  for (int i = 0; i < 8; ++i) z[i] = 0.0f;
  return z;
}

// 16B global -> LDS copy. Async (ASYNCcnt) on CDNA5 toolchains that expose the
// builtin; synchronous register bounce otherwise.
static __device__ __forceinline__ void cp_b128(const unsigned short* g,
                                               unsigned short* l) {
#ifdef USE_ASYNC_LDS
  __builtin_amdgcn_global_load_async_to_lds_b128(
      (__attribute__((address_space(1))) v4i_t*)const_cast<unsigned short*>(g),
      (__attribute__((address_space(3))) v4i_t*)l, 0, 0);
#else
  *(uint4*)l = *(const uint4*)g;
#endif
}
static __device__ __forceinline__ void wait_async_copies() {
#ifdef USE_ASYNC_LDS
  __builtin_amdgcn_s_wait_asynccnt(0);
#endif
}

// ------------------------------ small casts --------------------------------

__global__ void cast_f32_bf16_kernel(const float* __restrict__ src,
                                     unsigned short* __restrict__ dst, int n) {
  int i = blockIdx.x * blockDim.x + threadIdx.x;
  if (i < n) dst[i] = f2bf(src[i]);
}

// rows x sc  ->  rows x dc, zero-padded columns (for dt_proj K: 48 -> 64)
__global__ void cast_pad_kernel(const float* __restrict__ src,
                                unsigned short* __restrict__ dst,
                                int rows, int sc, int dc) {
  int i = blockIdx.x * blockDim.x + threadIdx.x;
  if (i >= rows * dc) return;
  int r = i / dc, c = i - r * dc;
  dst[i] = (c < sc) ? f2bf(src[r * sc + c]) : (unsigned short)0;
}

// first DT_RANK columns of x_dbl (ld=80) -> bf16 rows of DT_PAD, zero-padded
__global__ void cast_dtin_kernel(const float* __restrict__ xdbl,
                                 unsigned short* __restrict__ dtin, int total) {
  int i = blockIdx.x * blockDim.x + threadIdx.x;
  if (i >= total) return;
  int r = i >> 6;          // / DT_PAD
  int c = i & (DT_PAD - 1);
  dtin[i] = (c < DT_RANK) ? f2bf(xdbl[r * XDBL_C + c]) : (unsigned short)0;
}

// --------------------------- LayerNorm -> bf16 -----------------------------

__global__ __launch_bounds__(256) void ln_cast_kernel(
    const float* __restrict__ h_r, const float* __restrict__ h_i,
    const float* __restrict__ ln_w, const float* __restrict__ ln_b,
    unsigned short* __restrict__ hn) {
  __shared__ float s1[256], s2[256];
  const int m = blockIdx.x, g = blockIdx.y, tid = threadIdx.x;
  const float* src = (g ? h_i : h_r) + (long long)m * D_MODEL;
  float x0 = src[tid], x1 = src[tid + 256], x2 = src[tid + 512];
  s1[tid] = x0 + x1 + x2;
  s2[tid] = x0 * x0 + x1 * x1 + x2 * x2;
  __syncthreads();
  for (int off = 128; off > 0; off >>= 1) {
    if (tid < off) { s1[tid] += s1[tid + off]; s2[tid] += s2[tid + off]; }
    __syncthreads();
  }
  const float mean = s1[0] * (1.0f / D_MODEL);
  const float var  = s2[0] * (1.0f / D_MODEL) - mean * mean;
  const float rst  = rsqrtf(var + 1e-5f);
  unsigned short* dst = hn + ((long long)g * ROWS + m) * D_MODEL;
  const float* w = ln_w + g * D_MODEL;
  const float* b = ln_b + g * D_MODEL;
  for (int c = tid; c < D_MODEL; c += 256)
    dst[c] = f2bf((src[c] - mean) * rst * w[c] + b[c]);
}

// ------------------------- WMMA bf16 GEMM (C = A*B^T) ----------------------
// A: [M,K] bf16 row-major, B: [N,K] bf16 row-major, C: [M,N] (f32 or bf16).
// Batched over blockIdx.z. Block tile 128x128, BK=32, 8 waves, each wave a
// 64x32 region (4x2 wmma tiles). Double-buffered LDS; next K tile streams in
// via async global->LDS while the current tile's 8 WMMAs execute.
// Requirements: M % 128 == 0, K % 32 == 0. GUARD_N handles ragged N (N=80).

template <typename CT>
static __device__ __forceinline__ void store_c(CT* p, float v);
template <>
__device__ __forceinline__ void store_c<float>(float* p, float v) { *p = v; }
template <>
__device__ __forceinline__ void store_c<unsigned short>(unsigned short* p, float v) { *p = f2bf(v); }

template <typename CT, bool GUARD_N>
__global__ __launch_bounds__(256) void gemm_bf16_wmma(
    const unsigned short* __restrict__ A, const unsigned short* __restrict__ B,
    CT* __restrict__ C, int M, int N, int K,
    long long strideA, long long strideB, long long strideC) {
  constexpr int BM = 128, BN = 128, BK = 32, LS = 40;  // LS: padded LDS row
  __shared__ __attribute__((aligned(16))) unsigned short As[2][BM * LS];
  __shared__ __attribute__((aligned(16))) unsigned short Bs[2][BN * LS];

  const int g = blockIdx.z;
  const unsigned short* Ag = A + (long long)g * strideA;
  const unsigned short* Bg = B + (long long)g * strideB;
  CT* Cg = C + (long long)g * strideC;

  const int bm = blockIdx.y * BM;
  const int bn = blockIdx.x * BN;
  const int tid = threadIdx.x;
  const int lane = tid & 31;
  const int wid = tid >> 5;
  const int wm = (wid & 1) * 64;   // wave row offset in block tile
  const int wn = (wid >> 1) * 32;  // wave col offset in block tile

  // two 16B chunks per tile per thread
  const int row0 = tid >> 2,          col0 = (tid & 3) * 8;
  const int row1 = (tid + 256) >> 2,  col1 = (tid & 3) * 8;

  if (GUARD_N) {
    // zero both B buffers once; rows >= N are never overwritten afterwards
    uint4 zz = make_uint4(0u, 0u, 0u, 0u);
    uint4* bp = (uint4*)&Bs[0][0];
#pragma unroll
    for (int i = tid; i < 2 * BN * LS / 8; i += 256) bp[i] = zz;
  }
  __syncthreads();

  auto issue_tile = [&](int k0, int buf) {
    cp_b128(Ag + (long long)(bm + row0) * K + k0 + col0, &As[buf][row0 * LS + col0]);
    cp_b128(Ag + (long long)(bm + row1) * K + k0 + col1, &As[buf][row1 * LS + col1]);
    if (!GUARD_N || bn + row0 < N)
      cp_b128(Bg + (long long)(bn + row0) * K + k0 + col0, &Bs[buf][row0 * LS + col0]);
    if (!GUARD_N || bn + row1 < N)
      cp_b128(Bg + (long long)(bn + row1) * K + k0 + col1, &Bs[buf][row1 * LS + col1]);
  };

  v8f acc[4][2];
#pragma unroll
  for (int im = 0; im < 4; ++im)
#pragma unroll
    for (int jn = 0; jn < 2; ++jn) acc[im][jn] = v8f_zero();

  const int r = lane & 15;
  const int kbA = (lane >> 4) * 8;   // A frag: lanes 0-15 K{0..7,16..23}; +8
  const int kbB = (lane >> 4) * 8;   // B frag: lanes 0-15 K{0..7,16..23}; +8

  issue_tile(0, 0);
  wait_async_copies();
  __syncthreads();

  int buf = 0;
  for (int k0 = 0; k0 < K; k0 += BK) {
    if (k0 + BK < K) issue_tile(k0 + BK, buf ^ 1);  // overlap with compute

    union Frag { v16bf v; uint4 q[2]; } a[4], b[2];
    const unsigned short* Asb = As[buf];
    const unsigned short* Bsb = Bs[buf];
#pragma unroll
    for (int im = 0; im < 4; ++im) {
      const unsigned short* p = &Asb[(wm + im * 16 + r) * LS + kbA];
      a[im].q[0] = *(const uint4*)(p);
      a[im].q[1] = *(const uint4*)(p + 16);
    }
#pragma unroll
    for (int jn = 0; jn < 2; ++jn) {
      const unsigned short* p = &Bsb[(wn + jn * 16 + r) * LS + kbB];
      b[jn].q[0] = *(const uint4*)(p);
      b[jn].q[1] = *(const uint4*)(p + 16);
    }
#pragma unroll
    for (int im = 0; im < 4; ++im)
#pragma unroll
      for (int jn = 0; jn < 2; ++jn)
        acc[im][jn] = __builtin_amdgcn_wmma_f32_16x16x32_bf16(
            false, a[im].v, false, b[jn].v, (short)0, acc[im][jn], false, false);

    wait_async_copies();
    __syncthreads();
    buf ^= 1;
  }

  // ---- store: lane -> (n = lane&15, m = vgpr + 8*(lane>=16))
  const int ncol = lane & 15;
  const int mhalf = (lane >> 4) * 8;
#pragma unroll
  for (int im = 0; im < 4; ++im)
#pragma unroll
    for (int jn = 0; jn < 2; ++jn) {
      const int n = bn + wn + jn * 16 + ncol;
      if (!GUARD_N || n < N) {
#pragma unroll
        for (int rr = 0; rr < 8; ++rr) {
          const int m = bm + wm + im * 16 + mhalf + rr;
          store_c(&Cg[(long long)m * N + n], acc[im][jn][rr]);
        }
      }
    }
}

// ------------------- causal depthwise conv (k=4) + SiLU --------------------

__global__ __launch_bounds__(256) void conv_silu_kernel(
    const unsigned short* __restrict__ xz, const float* __restrict__ cw,
    const float* __restrict__ cb, float* __restrict__ u,
    unsigned short* __restrict__ ubf, int rev) {
  const int g = blockIdx.z;
  const int m = blockIdx.y;               // scan-order row: b*SEQ + t
  const int d = blockIdx.x * 256 + threadIdx.x;
  const int b = m >> 10, t = m & (SEQ - 1);
  const float* w = cw + (g * D_INNER + d) * D_CONV;
  float acc = cb[g * D_INNER + d];
#pragma unroll
  for (int i = 0; i < D_CONV; ++i) {
    const int st = t - (D_CONV - 1) + i;  // x[t-3+i] * w[i]
    if (st >= 0) {
      const int mt = rev ? (SEQ - 1 - st) : st;
      const long long xi =
          ((long long)(g * ROWS + b * SEQ + mt)) * (2 * D_INNER) + d;
      acc += bf2f(xz[xi]) * w[i];
    }
  }
  const float s = silu_f(acc);
  const long long oi = ((long long)(g * ROWS) + m) * D_INNER + d;
  u[oi] = s;
  ubf[oi] = f2bf(s);
}

// -------------------------- selective scan ---------------------------------
// One thread per (g,b,d) channel; h[16] and A[16] in VGPRs; B_t/C_t staged in
// double-buffered LDS (one barrier per step). softplus(dt_raw + bias) folded.

__global__ __launch_bounds__(256) void scan_kernel(
    const float* __restrict__ u, const float* __restrict__ dtraw,
    const float* __restrict__ xdbl, const float* __restrict__ A_log,
    const float* __restrict__ dt_bias, const float* __restrict__ Dp,
    float* __restrict__ yacc, int rev) {
  __shared__ float BC[2][2 * D_STATE];
  const int g = blockIdx.z, b = blockIdx.y;
  const int d = blockIdx.x * 256 + threadIdx.x;
  float A[D_STATE], h[D_STATE];
#pragma unroll
  for (int n = 0; n < D_STATE; ++n) {
    A[n] = -__expf(A_log[(g * D_INNER + d) * D_STATE + n]);
    h[n] = 0.0f;
  }
  const float bias = dt_bias[g * D_INNER + d];
  const float Dd = Dp[g * D_INNER + d];
  const long long rowbase = (long long)(g * ROWS + b * SEQ);

  for (int t = 0; t < SEQ; ++t) {
    const long long m = rowbase + t;
    const int slot = t & 1;
    if (threadIdx.x < 2 * D_STATE)
      BC[slot][threadIdx.x] = xdbl[m * XDBL_C + DT_RANK + threadIdx.x];
    __syncthreads();
    float dtv = dtraw[m * D_INNER + d] + bias;
    dtv = (dtv > 20.0f) ? dtv : log1pf(__expf(dtv));  // softplus
    const float uv = u[m * D_INNER + d];
    const float du = dtv * uv;
    float ys = 0.0f;
#pragma unroll
    for (int n = 0; n < D_STATE; ++n) {
      h[n] = __expf(dtv * A[n]) * h[n] + du * BC[slot][n];
      ys += h[n] * BC[slot][D_STATE + n];
    }
    const float yv = ys + uv * Dd;
    const int tout = rev ? (SEQ - 1 - t) : t;
    const long long oi = (rowbase + tout) * D_INNER + d;
    if (rev) yacc[oi] += yv; else yacc[oi] = yv;
  }
}

// --------------------------- gating: y * silu(z) ---------------------------

__global__ void gating_kernel(const float* __restrict__ yacc,
                              const unsigned short* __restrict__ xz,
                              unsigned short* __restrict__ ybf, int total) {
  int i = blockIdx.x * blockDim.x + threadIdx.x;
  if (i >= total) return;
  const int d = i % D_INNER;
  const int row = i / D_INNER;  // g*ROWS + m
  const float z = bf2f(xz[(long long)row * (2 * D_INNER) + D_INNER + d]);
  ybf[i] = f2bf(yacc[i] * silu_f(z));
}

// ---------------------------------------------------------------------------

extern "C" void kernel_launch(void* const* d_in, const int* in_sizes, int n_in,
                              void* d_out, int out_size, void* d_ws,
                              size_t ws_size, hipStream_t stream) {
  (void)in_sizes; (void)n_in; (void)out_size; (void)ws_size;

  const float* h_r  = (const float*)d_in[0];
  const float* h_i  = (const float*)d_in[1];
  const float* ln_w = (const float*)d_in[2];
  const float* ln_b = (const float*)d_in[3];
  const float* in_w = (const float*)d_in[4];
  const float* conv_w[2] = {(const float*)d_in[5],  (const float*)d_in[12]};
  const float* conv_b[2] = {(const float*)d_in[6],  (const float*)d_in[13]};
  const float* xp_w[2]   = {(const float*)d_in[7],  (const float*)d_in[14]};
  const float* dtp_w[2]  = {(const float*)d_in[8],  (const float*)d_in[15]};
  const float* dtp_b[2]  = {(const float*)d_in[9],  (const float*)d_in[16]};
  const float* A_log[2]  = {(const float*)d_in[10], (const float*)d_in[17]};
  const float* D_p[2]    = {(const float*)d_in[11], (const float*)d_in[18]};
  const float* out_w = (const float*)d_in[19];
  float* out = (float*)d_out;

  // ---- static workspace layout (256B aligned) ----
  char* ws = (char*)d_ws;
  size_t off = 0;
  auto carve = [&](size_t bytes) {
    void* p = ws + off;
    off = (off + bytes + 255) & ~(size_t)255;
    return p;
  };
  unsigned short* hn_bf   = (unsigned short*)carve((size_t)2 * ROWS * D_MODEL * 2);
  unsigned short* w_in_bf = (unsigned short*)carve((size_t)2 * 2 * D_INNER * D_MODEL * 2);
  unsigned short* w_out_bf= (unsigned short*)carve((size_t)2 * D_MODEL * D_INNER * 2);
  unsigned short* w_xp_bf = (unsigned short*)carve((size_t)2 * XDBL_C * D_INNER * 2);
  unsigned short* w_dtp_bf= (unsigned short*)carve((size_t)2 * D_INNER * DT_PAD * 2);
  unsigned short* xz_bf   = (unsigned short*)carve((size_t)2 * ROWS * 2 * D_INNER * 2);
  float*          u_f32   = (float*)carve((size_t)2 * ROWS * D_INNER * 4);
  unsigned short* u_bf    = (unsigned short*)carve((size_t)2 * ROWS * D_INNER * 2); // reused as y_bf
  float*          xdbl    = (float*)carve((size_t)2 * ROWS * XDBL_C * 4);
  unsigned short* dtin_bf = (unsigned short*)carve((size_t)2 * ROWS * DT_PAD * 2);
  float*          dt_f32  = (float*)carve((size_t)2 * ROWS * D_INNER * 4);
  float*          yacc    = (float*)carve((size_t)2 * ROWS * D_INNER * 4);

  // ---- weights -> bf16 ----
  {
    int n = 2 * 2 * D_INNER * D_MODEL;
    cast_f32_bf16_kernel<<<(n + 255) / 256, 256, 0, stream>>>(in_w, w_in_bf, n);
  }
  {
    int n = 2 * D_MODEL * D_INNER;
    cast_f32_bf16_kernel<<<(n + 255) / 256, 256, 0, stream>>>(out_w, w_out_bf, n);
  }

  // ---- LayerNorm -> bf16 activations ----
  ln_cast_kernel<<<dim3(ROWS, 2), 256, 0, stream>>>(h_r, h_i, ln_w, ln_b, hn_bf);

  // ---- in_proj: xz = hn @ in_w^T  (M=4096, N=3072, K=768) ----
  gemm_bf16_wmma<unsigned short, false>
      <<<dim3(2 * D_INNER / 128, ROWS / 128, 2), 256, 0, stream>>>(
      hn_bf, w_in_bf, xz_bf, ROWS, 2 * D_INNER, D_MODEL,
      (long long)ROWS * D_MODEL, (long long)2 * D_INNER * D_MODEL,
      (long long)ROWS * 2 * D_INNER);

  for (int dir = 0; dir < 2; ++dir) {
    {
      int n = 2 * XDBL_C * D_INNER;
      cast_f32_bf16_kernel<<<(n + 255) / 256, 256, 0, stream>>>(xp_w[dir], w_xp_bf, n);
    }
    {
      int n = 2 * D_INNER * DT_PAD;
      cast_pad_kernel<<<(n + 255) / 256, 256, 0, stream>>>(
          dtp_w[dir], w_dtp_bf, 2 * D_INNER, DT_RANK, DT_PAD);
    }
    // conv + SiLU (direction-flipped time for dir==1)
    conv_silu_kernel<<<dim3(D_INNER / 256, ROWS, 2), 256, 0, stream>>>(
        xz_bf, conv_w[dir], conv_b[dir], u_f32, u_bf, dir);
    // x_proj: x_dbl = u @ xp_w^T  (M=4096, N=80, K=1536) -- ragged N, guarded
    gemm_bf16_wmma<float, true><<<dim3(1, ROWS / 128, 2), 256, 0, stream>>>(
        u_bf, w_xp_bf, xdbl, ROWS, XDBL_C, D_INNER,
        (long long)ROWS * D_INNER, (long long)XDBL_C * D_INNER,
        (long long)ROWS * XDBL_C);
    // dt_proj input: pad K 48 -> 64
    {
      int n = 2 * ROWS * DT_PAD;
      cast_dtin_kernel<<<(n + 255) / 256, 256, 0, stream>>>(xdbl, dtin_bf, n);
    }
    // dt_proj: dt_raw = dtin @ dtp_w^T  (M=4096, N=1536, K=64)
    gemm_bf16_wmma<float, false>
        <<<dim3(D_INNER / 128, ROWS / 128, 2), 256, 0, stream>>>(
        dtin_bf, w_dtp_bf, dt_f32, ROWS, D_INNER, DT_PAD,
        (long long)ROWS * DT_PAD, (long long)D_INNER * DT_PAD,
        (long long)ROWS * D_INNER);
    // selective scan (softplus+bias folded); fwd writes, bwd adds flipped
    scan_kernel<<<dim3(D_INNER / 256, BATCH, 2), 256, 0, stream>>>(
        u_f32, dt_f32, xdbl, A_log[dir], dtp_b[dir], D_p[dir], yacc, dir);
  }

  // ---- gate with silu(z) (shared by fwd+bwd), reuse u_bf as y_bf ----
  {
    int n = 2 * ROWS * D_INNER;
    gating_kernel<<<(n + 255) / 256, 256, 0, stream>>>(yacc, xz_bf, u_bf, n);
  }

  // ---- out_proj: out = y @ out_w^T  (M=4096, N=768, K=1536) ----
  gemm_bf16_wmma<float, false>
      <<<dim3(D_MODEL / 128, ROWS / 128, 2), 256, 0, stream>>>(
      u_bf, w_out_bf, out, ROWS, D_MODEL, D_INNER,
      (long long)ROWS * D_INNER, (long long)D_MODEL * D_INNER,
      (long long)ROWS * D_MODEL);
}
